// MyRNN_53120155517494
// MI455X (gfx1250) — matrix operand order repaired and checked by reading, not compiled, not run on hardware
//
#include <hip/hip_runtime.h>
#include <cstddef>
#include <cstdint>

// ---------------------------------------------------------------------------
// Bidirectional GRU (shared weights), B=32, T=2048, IN=256, H=256, fp32 I/O.
//
// Pass 1 (gru_gx_kernel): gx[t][b][768] = x·W_ih^T + b_ih via
//   v_wmma_f32_16x16x32_bf16 (bf16 operands, fp32 accum).
// Pass 2 (gru_scan_kernel): 4 persistent workgroups = {dir 0,1} x {half 0,1}.
//   - 384x256 W_hh slice resident in LDS (bf16, pre-swizzled to the WMMA
//     B-operand lane layout).
//   - Per step: async-DMA the needed gx slice into LDS
//     (global_load_async_to_lds_b128 / s_wait_asynccnt) overlapping the WMMA
//     chain; h-half exchange through L2 (double-buffered) + atomic counters.
// ---------------------------------------------------------------------------

typedef __attribute__((ext_vector_type(16))) __bf16 v16bf;
typedef __attribute__((ext_vector_type(8)))  float  v8f;

#define TSTEPS 2048
#define BATCH  32
#define INSZ   256
#define HID    256
#define G3     768   // 3*HID gate rows

// workspace layout (bytes)
static constexpr size_t GX_BYTES     = (size_t)TSTEPS * BATCH * G3 * 4; // 201,326,592
static constexpr size_t ARRIVE_BYTES = 2 * TSTEPS * sizeof(int);        // 16,384

// LDS carve (bytes) for scan kernel
static constexpr int LDS_BW  = 0;               // 24*8*32 v16bf = 196,608
static constexpr int LDS_AS  = 196608;          // 2*8*32  v16bf =  16,384
static constexpr int LDS_GH  = 212992;          // 32*388 fp32   =  49,664 (padded stride)
static constexpr int LDS_GXS = 262656;          // 96 runs *128 fp32 = 49,152 (gx stage)
static constexpr int LDS_TOT = 311808;
static constexpr int GH_STRIDE = 388;           // pad: avoids 2-way bank conflict

__global__ void gru_init_kernel(int* arrive, float* hbuf) {
  int i = blockIdx.x * blockDim.x + threadIdx.x;
  if (i < 2 * TSTEPS) arrive[i] = 0;
  if (i < 2 * 2 * BATCH * HID) hbuf[i] = 0.0f;
}

// ---------------- Pass 1: gx = x @ W_ih^T + b_ih -> gx[t][b][768] ----------
__global__ __launch_bounds__(256) void gru_gx_kernel(
    const float* __restrict__ x, const float* __restrict__ W_ih,
    const float* __restrict__ b_ih, float* __restrict__ gx) {
  const int t    = blockIdx.x;
  const int wave = threadIdx.x >> 5;
  const int lane = threadIdx.x & 31;
  const int tile = blockIdx.y * 8 + wave;   // 0..95
  const int b0   = (tile & 1) * 16;
  const int g0   = (tile >> 1) * 16;
  const int lh   = lane >> 4;               // lane half
  const int lm   = lane & 15;

  const float* xrow = x    + ((size_t)(b0 + lm) * TSTEPS + t) * INSZ;
  const float* wrow = W_ih + (size_t)(g0 + lm) * INSZ;

  v8f acc = {};
  for (int kc = 0; kc < 8; ++kc) {
    v16bf a, bm;
    // A lane layout: M=lane%16; K = kc*32 + lh*16 + e
    const float* ap1 = xrow + kc * 32 + lh * 16;
    const float* ap2 = ap1 + 8;
#pragma unroll
    for (int e = 0; e < 8; ++e) { a[e] = (__bf16)ap1[e]; a[e + 8] = (__bf16)ap2[e]; }
    // B lane layout: N=lane%16; K = kc*32 + lh*16 + e
    const float* bp = wrow + kc * 32 + lh * 16;
#pragma unroll
    for (int e = 0; e < 16; ++e) bm[e] = (__bf16)bp[e];
    acc = __builtin_amdgcn_wmma_f32_16x16x32_bf16(false, a, false, bm,
                                                  (short)0, acc, false, false);
  }
  const float bias = b_ih[g0 + lm];
  // C layout: row M = v + 8*lh, col N = lane%16
  float* orow = gx + (size_t)t * (BATCH * G3) + (size_t)(b0 + lh * 8) * G3 + g0 + lm;
#pragma unroll
  for (int v = 0; v < 8; ++v) orow[(size_t)v * G3] = acc[v] + bias;
}

// ---------------- Pass 2: persistent recurrent scan ------------------------
__global__ __launch_bounds__(256) void gru_scan_kernel(
    const float* __restrict__ W_hh, const float* __restrict__ b_hh,
    const float* __restrict__ gx, float* __restrict__ out,
    float* hbuf, int* arrive) {
  extern __shared__ char smem[];
  __bf16* BwS = (__bf16*)(smem + LDS_BW);
  v16bf*  Bw  = (v16bf*)(smem + LDS_BW);
  __bf16* AsS = (__bf16*)(smem + LDS_AS);
  v16bf*  As  = (v16bf*)(smem + LDS_AS);
  float*  ghs = (float*)(smem + LDS_GH);
  float*  gxs = (float*)(smem + LDS_GXS);  // [run=b*3+gate][128]

  const int d    = blockIdx.x >> 1;   // direction
  const int hf   = blockIdx.x & 1;    // hidden half
  const int tid  = threadIdx.x;
  const int wave = tid >> 5;
  const int lane = tid & 31;
  const int lh   = lane >> 4;
  const int lm   = lane & 15;

  // Preload W_hh slice (384 gate-rows x 256 k) as bf16, pre-swizzled into the
  // WMMA B-operand lane layout: Bw[(ntile*8+kc)*32 + lane][e].
  for (int idx = tid; idx < 24 * 8 * 32; idx += 256) {
    int ntile = idx >> 8;
    int kc    = (idx >> 5) & 7;
    int ln    = idx & 31;
    int rloc  = ntile * 16 + (ln & 15);        // 0..383
    int gate  = rloc >> 7;
    int jj    = rloc & 127;
    int g     = gate * 256 + hf * 128 + jj;    // global gate row
    int kb    = kc * 32 + (ln >> 4) * 16;
    const float* wp = W_hh + (size_t)g * HID + kb;
#pragma unroll
    for (int e = 0; e < 16; ++e) BwS[idx * 16 + e] = (__bf16)wp[e];
  }

  float* hslot = hbuf + (size_t)d * 2 * (BATCH * HID);
  int*   arr   = arrive + d * TSTEPS;
  const size_t   HIDBASE  = (size_t)BATCH * TSTEPS * 512;  // hiddens offset
  const uint32_t gxs_base = (uint32_t)(uintptr_t)gxs;      // LDS byte offset

  __syncthreads();

  v8f acc[6];
  for (int s = 0; s < TSTEPS; ++s) {
    const int t = d ? (TSTEPS - 1 - s) : s;
    const float* hcur  = hslot + (s & 1) * (BATCH * HID);
    float*       hnext = hslot + ((s + 1) & 1) * (BATCH * HID);
    const float* gxt   = gx + (size_t)t * (BATCH * G3);

    // Kick off async DMA of this step's gx slice (only our 384 gate-rows) into
    // LDS; 12 b128 transfers per lane, tracked by ASYNCcnt. Overlaps with the
    // A-operand build and the WMMA chain below.
#pragma unroll
    for (int i = 0; i < 12; ++i) {
      int idx  = tid + 256 * i;       // 0..3071 float4 chunks
      int run  = idx >> 5;            // 0..95 : run = b*3 + gate
      int c    = idx & 31;            // float4 within run
      int b    = run / 3;
      int gate = run - b * 3;
      const float* gp = gxt + (size_t)b * G3 + gate * 256 + hf * 128 + c * 4;
      uint32_t loff = gxs_base + (uint32_t)((run * 128 + c * 4) * 4);
      asm volatile("global_load_async_to_lds_b128 %0, %1, off"
                   :: "v"(loff), "v"((uint64_t)(uintptr_t)gp) : "memory");
    }

    // Prefetch next step's gx slice toward L2 while this step computes.
    if (s + 1 < TSTEPS) {
      const int tn = d ? (TSTEPS - 2 - s) : (s + 1);
      const float* nx = gx + (size_t)tn * (BATCH * G3);
      __builtin_prefetch(nx + tid * 96, 0, 1);
      __builtin_prefetch(nx + tid * 96 + 32, 0, 1);
      __builtin_prefetch(nx + tid * 96 + 64, 0, 1);
    }

    // Build A-operand (bf16, WMMA lane layout) from the h exchange buffer.
    for (int idx = tid; idx < 2 * 8 * 32; idx += 256) {
      int mt   = idx >> 8;
      int kc   = (idx >> 5) & 7;
      int ln   = idx & 31;
      int b    = mt * 16 + (ln & 15);
      int koff = (ln >> 4) * 16;
      const float* hp = hcur + b * HID;
#pragma unroll
      for (int e = 0; e < 16; ++e) {
        int k = kc * 32 + e + koff;
        AsS[idx * 16 + e] = (__bf16)hp[k];
      }
    }
    __syncthreads();

    // gh = h @ Whh_slice^T : wave w owns ntiles 3w..3w+2, both batch tiles.
#pragma unroll
    for (int i = 0; i < 6; ++i) { v8f z = {}; acc[i] = z; }
    for (int kc = 0; kc < 8; ++kc) {
      v16bf a0 = As[(0 * 8 + kc) * 32 + lane];
      v16bf a1 = As[(1 * 8 + kc) * 32 + lane];
#pragma unroll
      for (int nt = 0; nt < 3; ++nt) {
        v16bf bb = Bw[((wave * 3 + nt) * 8 + kc) * 32 + lane];
        acc[nt * 2 + 0] = __builtin_amdgcn_wmma_f32_16x16x32_bf16(
            false, a0, false, bb, (short)0, acc[nt * 2 + 0], false, false);
        acc[nt * 2 + 1] = __builtin_amdgcn_wmma_f32_16x16x32_bf16(
            false, a1, false, bb, (short)0, acc[nt * 2 + 1], false, false);
      }
    }
    // Scatter gh tiles to LDS: row M = v + 8*lh (+16*mt), col = ntile*16+lm.
#pragma unroll
    for (int nt = 0; nt < 3; ++nt) {
      int col = (wave * 3 + nt) * 16 + lm;
#pragma unroll
      for (int mt = 0; mt < 2; ++mt) {
        int brow = mt * 16 + lh * 8;
#pragma unroll
        for (int v = 0; v < 8; ++v)
          ghs[(brow + v) * GH_STRIDE + col] = acc[nt * 2 + mt][v];
      }
    }
    // All async gx DMAs for this step must have landed before the barrier.
    asm volatile("s_wait_asynccnt 0x0" ::: "memory");
    __syncthreads();

    // Gate combine for this WG's 128 hidden units x 32 batch.
    for (int p = tid; p < BATCH * 128; p += 256) {
      int b  = p >> 7;
      int jj = p & 127;
      int j  = hf * 128 + jj;
      float hr = ghs[b * GH_STRIDE + jj]       + b_hh[0 * 256 + j];
      float hz = ghs[b * GH_STRIDE + 128 + jj] + b_hh[1 * 256 + j];
      float hn = ghs[b * GH_STRIDE + 256 + jj] + b_hh[2 * 256 + j];
      float xr = gxs[(b * 3 + 0) * 128 + jj];
      float xz = gxs[(b * 3 + 1) * 128 + jj];
      float xn = gxs[(b * 3 + 2) * 128 + jj];
      float r  = 1.0f / (1.0f + __expf(-(xr + hr)));
      float z  = 1.0f / (1.0f + __expf(-(xz + hz)));
      float n  = tanhf(xn + r * hn);
      float hnew = (1.0f - z) * n + z * hcur[b * HID + j];
      hnext[b * HID + j] = hnew;
      // outputs[b][t][d*256+j]
      out[(size_t)b * (TSTEPS * 512) + (size_t)t * 512 + d * 256 + j] = hnew;
      // hiddens[t][b][d*256+j]
      out[HIDBASE + ((size_t)t * BATCH + b) * 512 + d * 256 + j] = hnew;
    }

    // Cross-workgroup handshake with the partner half of this direction.
    __threadfence();
    __syncthreads();
    if (tid == 0) {
      __hip_atomic_fetch_add(&arr[s], 1, __ATOMIC_RELEASE, __HIP_MEMORY_SCOPE_AGENT);
      while (__hip_atomic_load(&arr[s], __ATOMIC_ACQUIRE, __HIP_MEMORY_SCOPE_AGENT) < 2) {}
      __threadfence();
    }
    __syncthreads();
  }
}

extern "C" void kernel_launch(void* const* d_in, const int* in_sizes, int n_in,
                              void* d_out, int out_size, void* d_ws, size_t ws_size,
                              hipStream_t stream) {
  const float* x    = (const float*)d_in[0];
  const float* W_ih = (const float*)d_in[1];
  const float* W_hh = (const float*)d_in[2];
  const float* b_ih = (const float*)d_in[3];
  const float* b_hh = (const float*)d_in[4];
  float* out = (float*)d_out;

  float* gx     = (float*)d_ws;                                   // 192 MB
  int*   arrive = (int*)((char*)d_ws + GX_BYTES);                 // 16 KB
  float* hbuf   = (float*)((char*)d_ws + GX_BYTES + ARRIVE_BYTES);// 128 KB

  // Re-zero control state every call (deterministic under graph replay).
  gru_init_kernel<<<128, 256, 0, stream>>>(arrive, hbuf);

  dim3 g1(TSTEPS, 12);
  gru_gx_kernel<<<g1, 256, 0, stream>>>(x, W_ih, b_ih, gx);

  hipFuncSetAttribute((const void*)gru_scan_kernel,
                      hipFuncAttributeMaxDynamicSharedMemorySize, LDS_TOT);
  gru_scan_kernel<<<4, 256, LDS_TOT, stream>>>(W_hh, b_hh, gx, out, hbuf, arrive);
}
